// TransformerEncoder_43602507989121
// MI455X (gfx1250) — compile-verified, matrix operand order checked by reading, not run on hardware
//
#include <hip/hip_runtime.h>
#include <hip/hip_bf16.h>

typedef __bf16 bf16;
typedef __attribute__((ext_vector_type(16))) __bf16 v16bf;
typedef __attribute__((ext_vector_type(8)))  __bf16 v8bf;
typedef __attribute__((ext_vector_type(8)))  float  v8f;

#define B_    128
#define S_    80
#define E_    1024
#define H_    16
#define D_    64
#define MLPD_ 4096
#define L_    6
#define NTOK  (B_*S_)

// ---------------------------------------------------------------------------
// CDNA5 async copy: 16B global -> LDS, tracked with ASYNCcnt (ISA §15.18.3).
// GVS mode: mem_addr = SGPR_base + VGPR_offset(i32) ; VDST = LDS byte address.
// ---------------------------------------------------------------------------
__device__ __forceinline__ void async_ld16(unsigned lds, unsigned long long base,
                                           unsigned voff) {
  asm volatile("global_load_async_to_lds_b128 %0, %1, %2"
               :: "v"(lds), "v"(voff), "s"(base) : "memory");
}
__device__ __forceinline__ void wait_async0() {
  asm volatile("s_wait_asynccnt 0" ::: "memory");
}
__device__ __forceinline__ unsigned lds_addr(const void* p) {
  return (unsigned)(unsigned long long)(uintptr_t)p;   // LDS aperture: low 32 bits
}

// ---------------------------------------------------------------------------
// WMMA helper: D = A(16x32 bf16) * B(32x16 bf16) + C(16x16 f32)
// ---------------------------------------------------------------------------
__device__ __forceinline__ v8f wmma_bf16(v8bf alo, v8bf ahi, v8bf blo, v8bf bhi, v8f c) {
  v16bf a = __builtin_shufflevector(alo, ahi, 0,1,2,3,4,5,6,7,8,9,10,11,12,13,14,15);
  v16bf b = __builtin_shufflevector(blo, bhi, 0,1,2,3,4,5,6,7,8,9,10,11,12,13,14,15);
  return __builtin_amdgcn_wmma_f32_16x16x32_bf16(false, a, false, b, (short)0, c, false, false);
}

// ---------------------------------------------------------------------------
// Tiled bf16 WMMA GEMM:  C[M,N] = act(A[M,K] @ W[K,N] + bias) + Rsd
// A is bf16 row-major (async-copied raw into LDS, double buffered);
// W is f32 row-major [K][N], staged through registers and converted to bf16.
// BM=BN=128, BK=32; 256 threads = 8 waves; wave tile 32x64 (2x4 WMMA tiles).
// ACT: 0=none, 1=sigmoid, 2=exact GELU.  OUT: 0=f32, 1=bf16, 2=both.
// ---------------------------------------------------------------------------
template<int ACT, bool HASBIAS, bool HASRES, int OUT>
__global__ __launch_bounds__(256)
void gemm_k(const bf16* __restrict__ A, const float* __restrict__ W,
            const float* __restrict__ bias, const float* __restrict__ Rsd,
            float* __restrict__ Cf, bf16* __restrict__ Cb,
            int M, int N, int K)
{
  __shared__ alignas(16) bf16 As[2][128][32];   // [m][k], raw async copy
  __shared__ alignas(16) bf16 Bs[2][128][32];   // [n][k] (transposed)
  const int tid   = threadIdx.x;
  const int wave  = tid >> 5;
  const int lane  = tid & 31;
  const int mhalf = lane >> 4;     // selects K-subblock per WMMA operand layout
  const int mrow  = lane & 15;
  const int bm = blockIdx.y * 128, bn = blockIdx.x * 128;
  const int wm = (wave & 3) * 32;
  const int wn = (wave >> 2) * 64;
  const unsigned long long abase =
      (unsigned long long)(uintptr_t)(A + (size_t)bm * K);

  const v8f vzero = {0.f,0.f,0.f,0.f,0.f,0.f,0.f,0.f};
  v8f acc[2][4];
  for (int mi = 0; mi < 2; ++mi)
    for (int ni = 0; ni < 4; ++ni) acc[mi][ni] = vzero;

  // async A-tile: 128x32 bf16 = 512 x 16B chunks; 2 per thread
  auto loadA_async = [&](int k0, int p) {
    #pragma unroll
    for (int j = 0; j < 2; ++j) {
      int c = tid + j * 256;
      int m = c >> 2, cc = c & 3;
      async_ld16(lds_addr(&As[p][m][cc * 8]), abase,
                 (unsigned)((m * K + k0 + cc * 8) * 2));
    }
  };
  // W tile: 32x128 f32, 16 values per thread (coalesced along n)
  auto loadW = [&](int k0, float* wreg) {
    #pragma unroll
    for (int t = 0; t < 16; ++t) {
      int i = tid + t * 256;
      wreg[t] = W[(size_t)(k0 + (i >> 7)) * N + bn + (i & 127)];
    }
  };
  auto storeW = [&](int p, const float* wreg) {
    #pragma unroll
    for (int t = 0; t < 16; ++t) {
      int i = tid + t * 256;
      Bs[p][i & 127][i >> 7] = (bf16)wreg[t];
    }
  };

  // prologue: tile 0
  loadA_async(0, 0);
  {
    float wreg[16];
    loadW(0, wreg);
    storeW(0, wreg);
  }
  wait_async0();
  __syncthreads();

  int p = 0;
  for (int k0 = 0; k0 < K; k0 += 32) {
    const bool hasnext = (k0 + 32 < K);
    float wreg[16];
    if (hasnext) {                 // prefetch next tile while WMMAs run
      loadA_async(k0 + 32, p ^ 1);
      loadW(k0 + 32, wreg);
    }

    // fragments: two aligned 16B LDS reads per operand per lane
    v8bf alo[2], ahi[2], blo[4], bhi[4];
    #pragma unroll
    for (int mi = 0; mi < 2; ++mi) {
      const v8bf* pa = (const v8bf*)&As[p][wm + mi*16 + mrow][0];
      alo[mi] = pa[mhalf]; ahi[mi] = pa[2 + mhalf];
    }
    #pragma unroll
    for (int ni = 0; ni < 4; ++ni) {
      const v8bf* pb = (const v8bf*)&Bs[p][wn + ni*16 + mrow][0];
      blo[ni] = pb[mhalf]; bhi[ni] = pb[2 + mhalf];
    }
    #pragma unroll
    for (int mi = 0; mi < 2; ++mi)
      #pragma unroll
      for (int ni = 0; ni < 4; ++ni)
        acc[mi][ni] = wmma_bf16(alo[mi], ahi[mi], blo[ni], bhi[ni], acc[mi][ni]);

    if (hasnext) {
      storeW(p ^ 1, wreg);
      wait_async0();               // this wave's async writes to As[p^1] landed
    }
    __syncthreads();               // single barrier per K-step (ping-pong)
    p ^= 1;
  }

  // Epilogue. C layout: VGPR r -> M = r + 8*(lane/16), N = lane%16.
  #pragma unroll
  for (int mi = 0; mi < 2; ++mi) {
    #pragma unroll
    for (int ni = 0; ni < 4; ++ni) {
      const int col = bn + wn + ni*16 + mrow;
      const float bv = HASBIAS ? bias[col] : 0.f;
      #pragma unroll
      for (int r = 0; r < 8; ++r) {
        const int row = bm + wm + mi*16 + r + 8*mhalf;
        float v = acc[mi][ni][r] + bv;
        if (ACT == 1)      v = 1.f / (1.f + expf(-v));
        else if (ACT == 2) v = 0.5f * v * (1.f + erff(v * 0.70710678118654752f));
        if (HASRES) v += Rsd[(size_t)row * N + col];
        if (OUT == 0 || OUT == 2) Cf[(size_t)row * N + col] = v;
        if (OUT == 1 || OUT == 2) Cb[(size_t)row * N + col] = (bf16)v;
      }
    }
  }
}

// ---------------------------------------------------------------------------
// Fused attention per (b, head): softmax(Q K^T / sqrt(D)) V.  All I/O bf16.
// 5 waves (160 threads); wave w owns q-rows [16w,16w+16).
// Q/K tiles async-copied to LDS; V loaded v8bf and transposed into LDS.
// reverse=1 indexes K/V rows at (S-1-k) (mirror attention).
// ---------------------------------------------------------------------------
__global__ __launch_bounds__(160)
void attn_k(const bf16* __restrict__ Q, const bf16* __restrict__ Kv,
            const bf16* __restrict__ Vv, bf16* __restrict__ O, int reverse)
{
  __shared__ alignas(16) bf16 Qs[80][64];
  __shared__ alignas(16) bf16 Ks[80][64];
  __shared__ alignas(16) bf16 Vt[64][96];   // transposed [d][k], k padded to 96
  __shared__ alignas(16) bf16 Ps[80][96];   // probabilities, cols padded to 96
  const int h = blockIdx.x, b = blockIdx.y;
  const int tid   = threadIdx.x;
  const int wave  = tid >> 5;
  const int lane  = tid & 31;
  const int mhalf = lane >> 4, mrow = lane & 15;
  const size_t base = ((size_t)b * S_) * E_ + (size_t)h * D_;
  const unsigned long long qbase = (unsigned long long)(uintptr_t)(Q + base);
  const unsigned long long kbase = (unsigned long long)(uintptr_t)(Kv + base);

  // 80x64 bf16 = 640 x 16B chunks per matrix; 4 per thread
  #pragma unroll
  for (int j = 0; j < 4; ++j) {
    int c = tid + j * 160;
    int m = c >> 3, cc = c & 7;
    int sk = reverse ? (79 - m) : m;
    async_ld16(lds_addr(&Qs[m][cc * 8]), qbase, (unsigned)((m  * E_ + cc * 8) * 2));
    async_ld16(lds_addr(&Ks[m][cc * 8]), kbase, (unsigned)((sk * E_ + cc * 8) * 2));
  }
  #pragma unroll
  for (int j = 0; j < 4; ++j) {     // V: load 8 elems, transpose-scatter to LDS
    int c = tid + j * 160;
    int m = c >> 3, cc = c & 7;
    int sk = reverse ? (79 - m) : m;
    v8bf vv = *(const v8bf*)&Vv[base + (size_t)sk * E_ + cc * 8];
    #pragma unroll
    for (int e = 0; e < 8; ++e) Vt[cc * 8 + e][m] = vv[e];
  }
  for (int i = tid; i < 64*16; i += 160) Vt[i >> 4][80 + (i & 15)] = (bf16)0.f;
  wait_async0();
  __syncthreads();

  // ---- scores: S[q,k] = sum_d Q[q,d] K[k,d], contract over d=64 (2 steps)
  const v8f vzero = {0.f,0.f,0.f,0.f,0.f,0.f,0.f,0.f};
  v8f sc[5];
  for (int ki = 0; ki < 5; ++ki) sc[ki] = vzero;
  const int qrow = wave*16 + mrow;
  for (int kk = 0; kk < 64; kk += 32) {
    const v8bf* pq = (const v8bf*)&Qs[qrow][kk];
    v8bf alo = pq[mhalf], ahi = pq[2 + mhalf];
    #pragma unroll
    for (int ki = 0; ki < 5; ++ki) {
      const v8bf* pk = (const v8bf*)&Ks[ki*16 + mrow][kk];
      sc[ki] = wmma_bf16(alo, ahi, pk[mhalf], pk[2 + mhalf], sc[ki]);
    }
  }

  // ---- row softmax (rows live in 16-lane half-groups; shfl_xor<16 stays inside)
  #pragma unroll
  for (int r = 0; r < 8; ++r) {
    float v[5]; float m = -3.4e38f;
    #pragma unroll
    for (int ki = 0; ki < 5; ++ki) { v[ki] = sc[ki][r] * 0.125f; m = fmaxf(m, v[ki]); }
    for (int o = 1; o < 16; o <<= 1) m = fmaxf(m, __shfl_xor(m, o, 32));
    float s = 0.f;
    #pragma unroll
    for (int ki = 0; ki < 5; ++ki) { v[ki] = expf(v[ki] - m); s += v[ki]; }
    for (int o = 1; o < 16; o <<= 1) s += __shfl_xor(s, o, 32);
    const float inv = 1.f / s;
    const int prow = wave*16 + r + 8*mhalf;
    #pragma unroll
    for (int ki = 0; ki < 5; ++ki) Ps[prow][ki*16 + mrow] = (bf16)(v[ki] * inv);
  }
  for (int i = tid; i < 80*16; i += 160) Ps[i >> 4][80 + (i & 15)] = (bf16)0.f;
  __syncthreads();

  // ---- O[q,d] = sum_k P[q,k] V[k,d], contract over k (padded 96, 3 steps)
  v8f oacc[4];
  for (int ni = 0; ni < 4; ++ni) oacc[ni] = vzero;
  for (int kk = 0; kk < 96; kk += 32) {
    const v8bf* pp = (const v8bf*)&Ps[wave*16 + mrow][kk];
    v8bf alo = pp[mhalf], ahi = pp[2 + mhalf];
    #pragma unroll
    for (int ni = 0; ni < 4; ++ni) {
      const v8bf* pv = (const v8bf*)&Vt[ni*16 + mrow][kk];
      oacc[ni] = wmma_bf16(alo, ahi, pv[mhalf], pv[2 + mhalf], oacc[ni]);
    }
  }
  #pragma unroll
  for (int ni = 0; ni < 4; ++ni) {
    const int dcol = ni*16 + mrow;
    #pragma unroll
    for (int r = 0; r < 8; ++r) {
      const int s = wave*16 + r + 8*mhalf;
      O[base + (size_t)s * E_ + dcol] = (bf16)oacc[ni][r];
    }
  }
}

// ---------------------------------------------------------------------------
// LayerNorm over E=1024: one block per token, 256 threads x float4 -> bf16 out.
// ---------------------------------------------------------------------------
__global__ __launch_bounds__(256)
void layernorm_k(const float* __restrict__ X, const float* __restrict__ g,
                 const float* __restrict__ bt, bf16* __restrict__ Y)
{
  const int row = blockIdx.x, tid = threadIdx.x;
  const int wave = tid >> 5, lane = tid & 31;
  __shared__ float red[8];
  float4 v = ((const float4*)(X + (size_t)row * E_))[tid];
  float s = v.x + v.y + v.z + v.w;
  for (int o = 16; o > 0; o >>= 1) s += __shfl_xor(s, o, 32);
  if (lane == 0) red[wave] = s;
  __syncthreads();
  float tot = 0.f;
  for (int i = 0; i < 8; ++i) tot += red[i];
  const float mean = tot * (1.f / (float)E_);
  const float dx = v.x - mean, dy = v.y - mean, dz = v.z - mean, dw = v.w - mean;
  float q = dx*dx + dy*dy + dz*dz + dw*dw;
  for (int o = 16; o > 0; o >>= 1) q += __shfl_xor(q, o, 32);
  __syncthreads();
  if (lane == 0) red[wave] = q;
  __syncthreads();
  float vt = 0.f;
  for (int i = 0; i < 8; ++i) vt += red[i];
  const float inv = rsqrtf(vt * (1.f / (float)E_) + 1e-5f);
  const float4 gg = ((const float4*)g)[tid];
  const float4 bb = ((const float4*)bt)[tid];
  bf16* y = Y + (size_t)row * E_ + tid * 4;
  y[0] = (bf16)(dx * inv * gg.x + bb.x);
  y[1] = (bf16)(dy * inv * gg.y + bb.y);
  y[2] = (bf16)(dz * inv * gg.z + bb.z);
  y[3] = (bf16)(dw * inv * gg.w + bb.w);
}

// ---------------------------------------------------------------------------
// Elementwise kernels
// ---------------------------------------------------------------------------
__global__ __launch_bounds__(256)
void embed_k(const int* __restrict__ x, const float* __restrict__ emb,
             float* __restrict__ xe)
{
  const int row = blockIdx.x;
  const int tok = x[row];
  const float4* src = (const float4*)(emb + (size_t)tok * E_);
  float4* dst = (float4*)(xe + (size_t)row * E_);
  dst[threadIdx.x] = src[threadIdx.x];
}

// sym features [x, x_rev, x*x_rev, |x-x_rev|] -> bf16 rows of 4096
// NOTE: |a-r| written as compare/select (NOT fabsf) to dodge a gfx1250 ISel
// crash on the packed sign-mask fabs idiom.
__global__ __launch_bounds__(256)
void symfeat_k(const float* __restrict__ xe, bf16* __restrict__ outp)
{
  const int row = blockIdx.x;
  const int b = row / S_, s = row % S_;
  const int rrow = b * S_ + (S_ - 1 - s);
  const int tid = threadIdx.x;
  const float4 a = ((const float4*)(xe + (size_t)row  * E_))[tid];
  const float4 r = ((const float4*)(xe + (size_t)rrow * E_))[tid];
  bf16* o = outp + (size_t)row * (4 * E_);
  const int c = tid * 4;
  o[c+0]=(bf16)a.x; o[c+1]=(bf16)a.y; o[c+2]=(bf16)a.z; o[c+3]=(bf16)a.w;
  bf16* o1 = o + E_;
  o1[c+0]=(bf16)r.x; o1[c+1]=(bf16)r.y; o1[c+2]=(bf16)r.z; o1[c+3]=(bf16)r.w;
  bf16* o2 = o + 2*E_;
  o2[c+0]=(bf16)(a.x*r.x); o2[c+1]=(bf16)(a.y*r.y);
  o2[c+2]=(bf16)(a.z*r.z); o2[c+3]=(bf16)(a.w*r.w);
  bf16* o3 = o + 3*E_;
  o3[c+0]=(bf16)(a.x > r.x ? a.x - r.x : r.x - a.x);
  o3[c+1]=(bf16)(a.y > r.y ? a.y - r.y : r.y - a.y);
  o3[c+2]=(bf16)(a.z > r.z ? a.z - r.z : r.z - a.z);
  o3[c+3]=(bf16)(a.w > r.w ? a.w - r.w : r.w - a.w);
}

// xe += gate * sym  (gate, sym bf16; xe f32)
__global__ __launch_bounds__(256)
void gatecomb_k(float* __restrict__ xe, const bf16* __restrict__ gate,
                const bf16* __restrict__ sym)
{
  const size_t i = (size_t)blockIdx.x * 256 + threadIdx.x;
  if (i >= (size_t)NTOK * E_) return;
  xe[i] += (float)gate[i] * (float)sym[i];
}

// concat [x, x_rev] -> bf16 rows of 2048
__global__ __launch_bounds__(256)
void catfeat_k(const float* __restrict__ xe, bf16* __restrict__ outp)
{
  const int row = blockIdx.x;
  const int b = row / S_, s = row % S_;
  const int rrow = b * S_ + (S_ - 1 - s);
  const int tid = threadIdx.x;
  const float4 a = ((const float4*)(xe + (size_t)row  * E_))[tid];
  const float4 r = ((const float4*)(xe + (size_t)rrow * E_))[tid];
  bf16* o = outp + (size_t)row * (2 * E_);
  const int c = tid * 4;
  o[c+0]=(bf16)a.x; o[c+1]=(bf16)a.y; o[c+2]=(bf16)a.z; o[c+3]=(bf16)a.w;
  bf16* o1 = o + E_;
  o1[c+0]=(bf16)r.x; o1[c+1]=(bf16)r.y; o1[c+2]=(bf16)r.z; o1[c+3]=(bf16)r.w;
}

__global__ __launch_bounds__(256)
void addpe_k(float* __restrict__ Hh)
{
  const size_t idx = (size_t)blockIdx.x * 256 + threadIdx.x;  // over B*S*E
  if (idx >= (size_t)NTOK * E_) return;
  const int d = (int)(idx & (E_ - 1));
  const int s = (int)((idx >> 10) % S_);
  const int i2 = d & ~1;
  const float freq = expf(-logf(10000.f) * (float)i2 / (float)E_);
  const float ang = (float)s * freq;
  Hh[idx] += (d & 1) ? cosf(ang) : sinf(ang);
}

// ---------------------------------------------------------------------------
// Host-side GEMM dispatch
// ---------------------------------------------------------------------------
static inline void gemm(hipStream_t st, int act,
                        const bf16* A, const float* W, const float* bias,
                        const float* Rsd, float* Cf, bf16* Cb,
                        int M, int N, int K)
{
  dim3 g(N / 128, M / 128), b(256);
  if (Rsd) {
    if (bias) gemm_k<0, true,  true, 0><<<g, b, 0, st>>>(A, W, bias, Rsd, Cf, Cb, M, N, K); // fc2
    else      gemm_k<0, false, true, 0><<<g, b, 0, st>>>(A, W, bias, Rsd, Cf, Cb, M, N, K); // o-proj
  } else if (act == 1) {
    gemm_k<1, true, false, 1><<<g, b, 0, st>>>(A, W, bias, Rsd, Cf, Cb, M, N, K);           // gate
  } else if (act == 2) {
    gemm_k<2, true, false, 1><<<g, b, 0, st>>>(A, W, bias, Rsd, Cf, Cb, M, N, K);           // fc1 GELU
  } else if (bias && Cf && Cb) {
    gemm_k<0, true, false, 2><<<g, b, 0, st>>>(A, W, bias, Rsd, Cf, Cb, M, N, K);           // h (both)
  } else if (bias) {
    gemm_k<0, true, false, 1><<<g, b, 0, st>>>(A, W, bias, Rsd, Cf, Cb, M, N, K);           // sym
  } else {
    gemm_k<0, false, false, 1><<<g, b, 0, st>>>(A, W, bias, Rsd, Cf, Cb, M, N, K);          // q/k/v
  }
}

// ---------------------------------------------------------------------------
extern "C" void kernel_launch(void* const* d_in, const int* in_sizes, int n_in,
                              void* d_out, int out_size, void* d_ws, size_t ws_size,
                              hipStream_t stream)
{
  (void)in_sizes; (void)n_in; (void)out_size; (void)ws_size;
  const int*   x      = (const int*)  d_in[0];
  const float* embed  = (const float*)d_in[2];
  const float* sym_w  = (const float*)d_in[3];
  const float* sym_b  = (const float*)d_in[4];
  const float* gate_w = (const float*)d_in[5];
  const float* gate_b = (const float*)d_in[6];
  const float* cat_w  = (const float*)d_in[7];
  const float* cat_b  = (const float*)d_in[8];
  const float* mir_wq = (const float*)d_in[9];
  const float* mir_wk = (const float*)d_in[10];
  const float* mir_wv = (const float*)d_in[11];
  const float* mir_wo = (const float*)d_in[12];
  const float* ln1_g  = (const float*)d_in[13];
  const float* ln1_b  = (const float*)d_in[14];
  const float* wq     = (const float*)d_in[15];
  const float* wk     = (const float*)d_in[16];
  const float* wv     = (const float*)d_in[17];
  const float* wo     = (const float*)d_in[18];
  const float* ln2_g  = (const float*)d_in[19];
  const float* ln2_b  = (const float*)d_in[20];
  const float* fc1_w  = (const float*)d_in[21];
  const float* fc1_b  = (const float*)d_in[22];
  const float* fc2_w  = (const float*)d_in[23];
  const float* fc2_b  = (const float*)d_in[24];
  float* out = (float*)d_out;

  // Workspace (~224 MB): 2 f32 E-wide buffers, 3 bf16 E-wide, 1 bf16 4E-wide.
  const size_t MBf = (size_t)NTOK * E_;
  float* buf0 = (float*)d_ws;          // x_emb / residual stream h (f32)
  float* buf1 = buf0 + MBf;            // h (f32) for mirror residual
  bf16*  bb0  = (bf16*)(buf1 + MBf);   // LN out / h(bf16)
  bf16*  bb1  = bb0 + MBf;             // sym / attn out
  bf16*  bb2  = bb1 + MBf;             // gate
  bf16*  bbig = bb2 + MBf;             // sym_in / cat_in / q,k,v / MLP mid
  bf16*  qb = bbig;
  bf16*  kb = bbig + MBf;
  bf16*  vb = bbig + 2 * MBf;

  // --- front end ---
  embed_k<<<NTOK, 256, 0, stream>>>(x, embed, buf0);
  symfeat_k<<<NTOK, 256, 0, stream>>>(buf0, bbig);
  gemm(stream, 0, bbig, sym_w, sym_b, nullptr, nullptr, bb1, NTOK, E_, 4 * E_);  // sym
  gemm(stream, 1, bb1, gate_w, gate_b, nullptr, nullptr, bb2, NTOK, E_, E_);     // gate
  gatecomb_k<<<(int)((MBf + 255) / 256), 256, 0, stream>>>(buf0, bb2, bb1);
  catfeat_k<<<NTOK, 256, 0, stream>>>(buf0, bbig);
  gemm(stream, 0, bbig, cat_w, cat_b, nullptr, buf1, bb0, NTOK, E_, 2 * E_);     // h (f32+bf16)

  // mirror attention: (h_rev @ W)[s] == (h @ W)[S-1-s] -> reverse inside attn
  gemm(stream, 0, bb0, mir_wq, nullptr, nullptr, nullptr, qb, NTOK, E_, E_);
  gemm(stream, 0, bb0, mir_wk, nullptr, nullptr, nullptr, kb, NTOK, E_, E_);
  gemm(stream, 0, bb0, mir_wv, nullptr, nullptr, nullptr, vb, NTOK, E_, E_);
  attn_k<<<dim3(H_, B_), 160, 0, stream>>>(qb, kb, vb, bb1, 1);
  gemm(stream, 0, bb1, mir_wo, nullptr, buf1, buf0, nullptr, NTOK, E_, E_);      // h += o@wo
  addpe_k<<<(int)(MBf / 256), 256, 0, stream>>>(buf0);

  // --- 6 encoder layers ---
  for (int l = 0; l < L_; ++l) {
    layernorm_k<<<NTOK, 256, 0, stream>>>(buf0, ln1_g + l * E_, ln1_b + l * E_, bb0);
    gemm(stream, 0, bb0, wq + (size_t)l * E_ * E_, nullptr, nullptr, nullptr, qb, NTOK, E_, E_);
    gemm(stream, 0, bb0, wk + (size_t)l * E_ * E_, nullptr, nullptr, nullptr, kb, NTOK, E_, E_);
    gemm(stream, 0, bb0, wv + (size_t)l * E_ * E_, nullptr, nullptr, nullptr, vb, NTOK, E_, E_);
    attn_k<<<dim3(H_, B_), 160, 0, stream>>>(qb, kb, vb, bb1, 0);
    gemm(stream, 0, bb1, wo + (size_t)l * E_ * E_, nullptr, buf0, buf0, nullptr, NTOK, E_, E_);
    layernorm_k<<<NTOK, 256, 0, stream>>>(buf0, ln2_g + l * E_, ln2_b + l * E_, bb0);
    gemm(stream, 2, bb0, fc1_w + (size_t)l * E_ * MLPD_, fc1_b + (size_t)l * MLPD_,
         nullptr, nullptr, bbig, NTOK, MLPD_, E_);                               // GELU
    float* Cdst = (l == L_ - 1) ? out : buf0;
    gemm(stream, 0, bbig, fc2_w + (size_t)l * MLPD_ * E_, fc2_b + (size_t)l * E_,
         buf0, Cdst, nullptr, NTOK, E_, MLPD_);                                  // + residual
  }
}
